// MultiChannelLinear_57664230916556
// MI455X (gfx1250) — matrix operand order repaired and checked by reading, not compiled, not run on hardware
//
#include <hip/hip_runtime.h>

typedef _Float16 v16h __attribute__((ext_vector_type(16)));
typedef _Float16 v8h  __attribute__((ext_vector_type(8)));
typedef _Float16 v4h  __attribute__((ext_vector_type(4)));
typedef float    v8f  __attribute__((ext_vector_type(8)));
typedef float    v4f  __attribute__((ext_vector_type(4)));

namespace {
constexpr int kN   = 32768;
constexpr int kC   = 8;
constexpr int kIn  = 512;
constexpr int kOut = 512;

// Block tile: 256(M) x 128(N), K-step 32. 8 waves (wave32) in a 4(M) x 2(N) grid,
// each wave computes a 64x64 tile as 4x4 WMMA accumulators (v8f each).
constexpr int BM = 256;
constexpr int BN = 128;
constexpr int BK = 32;
constexpr int LDA = 40;   // halves per LDS row (32 padded to 40 -> 80B stride, conflict-free b128)
constexpr int LDB = 40;
constexpr int WM = 64;
constexpr int WN = 64;
constexpr int THREADS = 256;
constexpr int ABUF = BM * LDA;   // halves per A buffer (10240)
constexpr int BBUF = BN * LDB;   // halves per B buffer (5120)
}

__global__ __launch_bounds__(THREADS)
void mcl_wmma_f16(const float* __restrict__ x,
                  const float* __restrict__ W,
                  const float* __restrict__ bias,
                  float* __restrict__ out)
{
  // Double-buffered staging: one barrier per K-step.
  __shared__ _Float16 As[2 * ABUF];   // 40960 B
  __shared__ _Float16 Bs[2 * BBUF];   // 20480 B

  const int n0 = blockIdx.x * BN;     // OUT tile (fastest -> L2 reuse of x tile)
  const int c  = blockIdx.y;          // channel
  const int m0 = blockIdx.z * BM;     // batch tile

  const int t    = threadIdx.x;
  const int lane = t & 31;
  const int wave = t >> 5;
  const int wm   = (wave & 3) * WM;
  const int wn   = (wave >> 2) * WN;
  const int half = lane >> 4;         // K-half select per WMMA operand layout
  const int l15  = lane & 15;

  // Uniform 64-bit bases; per-thread offsets kept in 32-bit ints so the
  // compiler can use SGPR-base + VGPR-offset (GVS) addressing.
  const float* xA = x + ((size_t)m0 * kC + c) * kIn;        // x[m0+row][c][.]
  const float* Wg = W + ((size_t)c * kOut + n0) * kIn;      // W[c][n0+row][.]

  // Per-thread load assignment (fixed row/col per j), 32-bit element offsets.
  int a_off[8];
  int b_off[4];
#pragma unroll
  for (int j = 0; j < 8; ++j) {
    const int idx4 = j * THREADS + t;
    a_off[j] = (idx4 >> 3) * (kC * kIn) + (idx4 & 7) * 4;   // row*4096 + col
  }
#pragma unroll
  for (int j = 0; j < 4; ++j) {
    const int idx4 = j * THREADS + t;
    b_off[j] = (idx4 >> 3) * kIn + (idx4 & 7) * 4;          // row*512 + col
  }

  v4f a_reg[8];   // 256x32 f32 tile / 256 threads = 8 float4
  v4f b_reg[4];   // 128x32 f32 tile / 256 threads = 4 float4

  auto load_tile = [&](int k0) {
#pragma unroll
    for (int j = 0; j < 8; ++j)
      a_reg[j] = *(const v4f*)(xA + (a_off[j] + k0));
#pragma unroll
    for (int j = 0; j < 4; ++j)
      b_reg[j] = *(const v4f*)(Wg + (b_off[j] + k0));
  };

  auto stage_tile = [&](int buf) {
    _Float16* Ad = As + buf * ABUF;
    _Float16* Bd = Bs + buf * BBUF;
#pragma unroll
    for (int j = 0; j < 8; ++j) {
      const int idx4 = j * THREADS + t;
      const int row  = idx4 >> 3;
      const int col  = (idx4 & 7) * 4;
      const v4f v = a_reg[j];
      v4h h;
      h[0] = (_Float16)v[0]; h[1] = (_Float16)v[1];
      h[2] = (_Float16)v[2]; h[3] = (_Float16)v[3];
      *(v4h*)(&Ad[row * LDA + col]) = h;
    }
#pragma unroll
    for (int j = 0; j < 4; ++j) {
      const int idx4 = j * THREADS + t;
      const int row  = idx4 >> 3;
      const int col  = (idx4 & 7) * 4;
      const v4f v = b_reg[j];
      v4h h;
      h[0] = (_Float16)v[0]; h[1] = (_Float16)v[1];
      h[2] = (_Float16)v[2]; h[3] = (_Float16)v[3];
      *(v4h*)(&Bd[row * LDB + col]) = h;
    }
  };

  v8f acc[4][4];
#pragma unroll
  for (int i = 0; i < 4; ++i)
#pragma unroll
    for (int j = 0; j < 4; ++j)
      acc[i][j] = {};

  load_tile(0);
  stage_tile(0);
  __syncthreads();

  constexpr int KT = kIn / BK;   // 16
  for (int kt = 0; kt < KT; ++kt) {
    const int cur = kt & 1;
    if (kt + 1 < KT) load_tile((kt + 1) * BK);   // global prefetch into registers

    const _Float16* Ab = As + cur * ABUF;
    const _Float16* Bb = Bs + cur * BBUF;

    // A fragment (16x32 f16): lanes 0-15 hold K 0-7 & 16-23, lanes 16-31 hold K 8-15 & 24-31
    v16h af[4];
#pragma unroll
    for (int tm = 0; tm < 4; ++tm) {
      const _Float16* p = &Ab[(wm + tm * 16 + l15) * LDA + half * 8];
      v8h lo = *(const v8h*)p;
      v8h hi = *(const v8h*)(p + 16);
      af[tm] = __builtin_shufflevector(lo, hi, 0,1,2,3,4,5,6,7,8,9,10,11,12,13,14,15);
    }
    // B fragment (32x16 f16): lanes 0-15 hold K 0-7 & 16-23 of column N, lanes 16-31 hold K 8-15 & 24-31
    v16h bf[4];
#pragma unroll
    for (int tn = 0; tn < 4; ++tn) {
      const _Float16* p = &Bb[(wn + tn * 16 + l15) * LDB + half * 8];
      v8h lo = *(const v8h*)p;
      v8h hi = *(const v8h*)(p + 16);
      bf[tn] = __builtin_shufflevector(lo, hi, 0,1,2,3,4,5,6,7,8,9,10,11,12,13,14,15);
    }

#pragma unroll
    for (int tm = 0; tm < 4; ++tm)
#pragma unroll
      for (int tn = 0; tn < 4; ++tn)
        acc[tm][tn] = __builtin_amdgcn_wmma_f32_16x16x32_f16(
            /*neg_a=*/false, af[tm], /*neg_b=*/false, bf[tn],
            /*c_mod=*/(short)0, acc[tm][tn],
            /*reuse_a=*/false, /*reuse_b=*/false);

    if (kt + 1 < KT) {
      stage_tile(cur ^ 1);   // convert + ds_store next buffer; overlaps WMMA pipe
      __syncthreads();
    }
  }

  // Epilogue: D layout — VGPR r: lanes 0-15 -> M=r, N=lane; lanes 16-31 -> M=8+r, N=lane-16
  float* outb = out + ((size_t)m0 * kC + c) * kOut;   // out[m0+row][c][.]
#pragma unroll
  for (int tn = 0; tn < 4; ++tn) {
    const int col = n0 + wn + tn * 16 + l15;
    const float bv = bias[c * kOut + col];
#pragma unroll
    for (int tm = 0; tm < 4; ++tm) {
      const int rbase = wm + tm * 16 + half * 8;      // row within block tile
#pragma unroll
      for (int r = 0; r < 8; ++r) {
        const int off = (rbase + r) * (kC * kOut) + col;   // < 2^31, 32-bit
        outb[off] = acc[tm][tn][r] + bv;
      }
    }
  }
}

extern "C" void kernel_launch(void* const* d_in, const int* in_sizes, int n_in,
                              void* d_out, int out_size, void* d_ws, size_t ws_size,
                              hipStream_t stream) {
  const float* x = (const float*)d_in[0];
  const float* W = (const float*)d_in[1];
  const float* b = (const float*)d_in[2];
  float* out     = (float*)d_out;

  dim3 grid(kOut / BN, kC, kN / BM);   // (4, 8, 128); OUT-tile fastest for x reuse in L2
  dim3 block(THREADS);
  hipLaunchKernelGGL(mcl_wmma_f16, grid, block, 0, stream, x, W, b, out);
}
